// PointNet2Segment_51599737094351
// MI455X (gfx1250) — matrix operand order repaired and checked by reading, not compiled, not run on hardware
//
#include <hip/hip_runtime.h>

typedef __attribute__((ext_vector_type(16))) _Float16 v16h;
typedef __attribute__((ext_vector_type(8)))  float    v8f;
typedef __attribute__((ext_vector_type(4)))  unsigned int u32x4;
typedef __attribute__((ext_vector_type(8)))  int      i32x8;
typedef __attribute__((ext_vector_type(4)))  int      i32x4;

#define NB   8
#define NP   4096
#define NS0  2048
#define NKN  64
#define R0SQ 0.04f

// ---------------------------------------------------------------------------
// WMMA fragment helpers (CDNA5 ISA 7.12.2 VGPR layouts, wave32)
// ---------------------------------------------------------------------------

// A fragment: 16x32 (MxK) f16 tile from row-major buf[(row0+m)*ld + k0 + k].
// lane L: M = L&15 ; lanes>=16 use K offset +8.
// half h -> K = (h&7) + (h&8 ? 16 : 0); both 8-half groups are contiguous
// in memory, so this lowers to 2x ds_load_b128 per lane.
__device__ __forceinline__ v16h frag_a(const _Float16* buf, int row0, int k0, int ld) {
  const int lane = threadIdx.x & 31;
  const int m    = lane & 15;
  const int koff = (lane & 16) ? 8 : 0;
  const int base = (row0 + m) * ld + k0 + koff;
  v16h a;
#pragma unroll
  for (int h = 0; h < 16; ++h) {
    const int k = (h & 7) + ((h & 8) << 1);
    a[h] = buf[base + k];
  }
  return a;
}

// B fragment from pre-packed fragment-major weights:
// packed[((kt*Nt + nt)*32 + lane)*16 + h]  ->  one 32B contiguous load per lane.
__device__ __forceinline__ v16h frag_bp(const _Float16* packed, int kt, int nt, int Nt) {
  const int lane = threadIdx.x & 31;
  return *(const v16h*)(packed + (((kt * Nt + nt) << 5) + lane) * 16);
}

__device__ __forceinline__ v8f wmma_f16(v16h a, v16h b, v8f c) {
  return __builtin_amdgcn_wmma_f32_16x16x32_f16(false, a, false, b, (short)0, c,
                                                false, false);
}

// ---------------------------------------------------------------------------
// Kernel 0: pack weights into f16 B-fragment-major layout (with K/N padding)
// ---------------------------------------------------------------------------
__device__ void pack_b(const float* __restrict__ src, _Float16* __restrict__ dst,
                       int Kt, int Nt, int Ksrc, int Nsrc, int t) {
  const int total = Kt * Nt * 512;
  for (int i = t; i < total; i += 256) {
    const int blk  = i >> 9;
    const int w    = i & 511;
    const int lane = w >> 4, h = w & 15;
    const int kt = blk / Nt, nt = blk % Nt;
    const int k = kt * 32 + ((lane & 16) ? 8 : 0) + (h & 7) + ((h & 8) << 1);   // B-layout K index
    const int n = nt * 16 + (lane & 15);                  // B-layout N index
    const float v = (k < Ksrc && n < Nsrc) ? src[k * Nsrc + n] : 0.f;
    dst[i] = (_Float16)v;
  }
}

__global__ void convert_weights_kernel(const float* __restrict__ W_sa0a,
                                       const float* __restrict__ W_sa0b,
                                       const float* __restrict__ W_fp0,
                                       const float* __restrict__ W_l0,
                                       const float* __restrict__ W_l1,
                                       _Float16* __restrict__ Wa,
                                       _Float16* __restrict__ Wb,
                                       _Float16* __restrict__ Wf,
                                       _Float16* __restrict__ Wl0h,
                                       _Float16* __restrict__ Wl1h) {
  const int t = threadIdx.x;
  pack_b(W_sa0a, Wa,   1, 4, 3,   64, t);
  pack_b(W_sa0b, Wb,   2, 8, 64, 128, t);
  pack_b(W_fp0,  Wf,   4, 4, 128, 64, t);
  pack_b(W_l0,   Wl0h, 2, 4, 64,  64, t);
  pack_b(W_l1,   Wl1h, 2, 1, 64,  10, t);
}

// ---------------------------------------------------------------------------
// Kernel 1: deterministic FPS, one block (1024 thr, 32 waves) per cloud.
// Points and running min-distance live entirely in VGPRs (4 per thread);
// fused min-update + argmax pass; 2 barriers per iteration.
// ---------------------------------------------------------------------------
__global__ __launch_bounds__(1024) void fps_kernel(const float* __restrict__ pos,
                                                   float* __restrict__ cpos) {
  __shared__ float wv[32];
  __shared__ int   wi[32];
  __shared__ float sb[3];

  const int t     = threadIdx.x;
  const int cloud = blockIdx.x;
  const float* cp = pos + (size_t)cloud * NP * 3;
  float* cpo      = cpos + (size_t)cloud * NS0 * 3;

  // register-resident chunk: points t*4 .. t*4+3
  float lx[4], ly[4], lz[4], md[4];
#pragma unroll
  for (int i = 0; i < 4; ++i) {
    const int j = t * 4 + i;
    lx[i] = cp[j * 3 + 0];
    ly[i] = cp[j * 3 + 1];
    lz[i] = cp[j * 3 + 2];
    md[i] = __builtin_inff();
  }

  float bx = cp[0], by = cp[1], bz = cp[2];   // current furthest point = pos[0]
  if (t == 0) { cpo[0] = bx; cpo[1] = by; cpo[2] = bz; }

  for (int it = 1; it < NS0; ++it) {
    // fused: md = min(md, d2(point, best)) and local argmax of md
    float v = -1.f; int ix = 0;
#pragma unroll
    for (int i = 0; i < 4; ++i) {
      const float dx = lx[i] - bx, dy = ly[i] - by, dz = lz[i] - bz;
      const float d = dx * dx + dy * dy + dz * dz;
      if (d < md[i]) md[i] = d;
      if (md[i] > v) { v = md[i]; ix = t * 4 + i; }   // ascending i: first max kept
    }
#pragma unroll
    for (int off = 16; off; off >>= 1) {
      float ov = __shfl_xor(v, off, 32);
      int   oi = __shfl_xor(ix, off, 32);
      if (ov > v || (ov == v && oi < ix)) { v = ov; ix = oi; }
    }
    if ((t & 31) == 0) { wv[t >> 5] = v; wi[t >> 5] = ix; }
    __syncthreads();
    if (t < 32) {                       // wave 0 reduces the 32 wave partials
      float v2 = wv[t]; int ix2 = wi[t];
#pragma unroll
      for (int off = 16; off; off >>= 1) {
        float ov = __shfl_xor(v2, off, 32);
        int   oi = __shfl_xor(ix2, off, 32);
        if (ov > v2 || (ov == v2 && oi < ix2)) { v2 = ov; ix2 = oi; }
      }
      if (t == 0) {
        const int bi = ix2;
        const float nbx = cp[bi * 3 + 0], nby = cp[bi * 3 + 1], nbz = cp[bi * 3 + 2];
        cpo[it * 3 + 0] = nbx; cpo[it * 3 + 1] = nby; cpo[it * 3 + 2] = nbz;
        sb[0] = nbx; sb[1] = nby; sb[2] = nbz;
      }
    }
    __syncthreads();
    bx = sb[0]; by = sb[1]; bz = sb[2];
  }
}

// ---------------------------------------------------------------------------
// Kernel 2: exact top-64 ball query via merge of per-thread sorted runs.
// ---------------------------------------------------------------------------
__global__ void ballquery_kernel(const float* __restrict__ pos,
                                 const float* __restrict__ cpos,
                                 int* __restrict__ nidx, float* __restrict__ nd2) {
  __shared__ float svals[256 * 17];   // stride 17: bank-spread
  __shared__ int   sidx [256 * 17];
  __shared__ float wv[8];
  __shared__ int   wi[8];
  __shared__ int   sBest;

  const int t     = threadIdx.x;
  const int c     = blockIdx.x;
  const int cloud = c / NS0;
  const float* cp = pos + (size_t)cloud * NP * 3;
  const float cx = cpos[c * 3 + 0], cy = cpos[c * 3 + 1], cz = cpos[c * 3 + 2];
  const int base = t * 17;

  for (int i = 0; i < 16; ++i) {
    const int j = t + (i << 8);
    const float dx = cp[j * 3 + 0] - cx, dy = cp[j * 3 + 1] - cy, dz = cp[j * 3 + 2] - cz;
    const float d = dx * dx + dy * dy + dz * dz;
    int p = i;
    while (p > 0 && svals[base + p - 1] > d) {
      svals[base + p] = svals[base + p - 1];
      sidx [base + p] = sidx [base + p - 1];
      --p;
    }
    svals[base + p] = d;
    sidx [base + p] = j;
  }

  int ptr = 0;
  for (int e = 0; e < NKN; ++e) {
    float v = (ptr < 16) ? svals[base + ptr] : __builtin_inff();
    int  ix = (ptr < 16) ? sidx [base + ptr] : 0x7fffffff;
#pragma unroll
    for (int off = 16; off; off >>= 1) {
      float ov = __shfl_xor(v, off, 32);
      int   oi = __shfl_xor(ix, off, 32);
      if (ov < v || (ov == v && oi < ix)) { v = ov; ix = oi; }
    }
    if ((t & 31) == 0) { wv[t >> 5] = v; wi[t >> 5] = ix; }
    __syncthreads();
    if (t == 0) {
      float bv = wv[0]; int bi = wi[0];
#pragma unroll
      for (int w = 1; w < 8; ++w)
        if (wv[w] < bv || (wv[w] == bv && wi[w] < bi)) { bv = wv[w]; bi = wi[w]; }
      nidx[(size_t)c * NKN + e] = bi;
      nd2 [(size_t)c * NKN + e] = bv;
      sBest = bi;
    }
    __syncthreads();
    if (t == (sBest & 255)) ++ptr;
  }
}

// ---------------------------------------------------------------------------
// Kernel 3: SA0 per-edge MLP (3->64->128) + masked max-pool, WMMA.
// ---------------------------------------------------------------------------
__global__ void sa0_mlp_kernel(const float* __restrict__ pos,
                               const float* __restrict__ cpos,
                               const int* __restrict__ nidx,
                               const float* __restrict__ nd2,
                               const _Float16* __restrict__ Wa,
                               const _Float16* __restrict__ Wb,
                               const float* __restrict__ ba,
                               const float* __restrict__ bb,
                               float* __restrict__ x0) {
  __shared__ _Float16 sA[64 * 32];
  __shared__ _Float16 sH[64 * 64];
  __shared__ float    sG[64 * 128];
  __shared__ float    sValid[64];

  const int t     = threadIdx.x;
  const int c     = blockIdx.x;
  const int cloud = c / NS0;
  const float cx = cpos[c * 3 + 0], cy = cpos[c * 3 + 1], cz = cpos[c * 3 + 2];

  if (t < 64) {
    const int ni    = nidx[(size_t)c * NKN + t];
    const float* pp = pos + ((size_t)cloud * NP + ni) * 3;
    _Float16* row = sA + t * 32;
    row[0] = (_Float16)(pp[0] - cx);
    row[1] = (_Float16)(pp[1] - cy);
    row[2] = (_Float16)(pp[2] - cz);
#pragma unroll
    for (int k = 3; k < 32; ++k) row[k] = (_Float16)0.f;
    sValid[t] = (nd2[(size_t)c * NKN + t] <= R0SQ) ? 1.f : 0.f;
  }
  __syncthreads();

  const int wave = t >> 5, lane = t & 31;
  const int n = lane & 15, mb = (lane & 16) ? 8 : 0;

  for (int tt = wave; tt < 16; tt += 8) {
    const int mt = tt & 3, nt = tt >> 2;
    v16h a = frag_a(sA, mt * 16, 0, 32);
    v16h b = frag_bp(Wa, 0, nt, 4);
    v8f acc = {};
    acc = wmma_f16(a, b, acc);
    const float bias = ba[nt * 16 + n];
#pragma unroll
    for (int i = 0; i < 8; ++i) {
      float v = acc[i] + bias;
      v = v > 0.f ? v : 0.f;
      sH[(mt * 16 + mb + i) * 64 + nt * 16 + n] = (_Float16)v;
    }
  }
  __syncthreads();

  for (int tt = wave; tt < 32; tt += 8) {
    const int mt = tt & 3, nt = tt >> 2;
    v8f acc = {};
#pragma unroll
    for (int kk = 0; kk < 2; ++kk) {
      v16h a = frag_a(sH, mt * 16, kk * 32, 64);
      v16h b = frag_bp(Wb, kk, nt, 8);
      acc = wmma_f16(a, b, acc);
    }
    const float bias = bb[nt * 16 + n];
#pragma unroll
    for (int i = 0; i < 8; ++i) {
      float v = acc[i] + bias;
      v = v > 0.f ? v : 0.f;
      sG[(mt * 16 + mb + i) * 128 + nt * 16 + n] = v;
    }
  }
  __syncthreads();

  if (t < 128) {
    float m = -__builtin_inff();
#pragma unroll 8
    for (int e = 0; e < NKN; ++e)
      if (sValid[e] > 0.f) {
        float v = sG[e * 128 + t];
        m = v > m ? v : m;
      }
    x0[(size_t)c * 128 + t] = m;
  }
}

// ---------------------------------------------------------------------------
// Kernel 4: knn_interpolate (k=3, inverse-d2); writes up0 directly as f16.
// ---------------------------------------------------------------------------
__global__ void knn_interp_kernel(const float* __restrict__ pos,
                                  const float* __restrict__ cpos,
                                  const float* __restrict__ x0,
                                  _Float16* __restrict__ up0h) {
  __shared__ float scx[NS0], scy[NS0], scz[NS0];

  const int t      = threadIdx.x;
  const size_t gid = (size_t)blockIdx.x * 256 + t;
  const int cloud  = (int)(gid / NP);
  const size_t cb  = (size_t)cloud * NS0;

  for (int i = t; i < NS0; i += 256) {
    const float* cp = cpos + (cb + i) * 3;
    scx[i] = cp[0]; scy[i] = cp[1]; scz[i] = cp[2];
  }
  __syncthreads();

  const float px = pos[gid * 3 + 0], py = pos[gid * 3 + 1], pz = pos[gid * 3 + 2];
  float d0 = __builtin_inff(), d1 = __builtin_inff(), d2v = __builtin_inff();
  int i0 = 0, i1 = 0, i2 = 0;
  for (int s = 0; s < NS0; ++s) {
    const float dx = px - scx[s], dy = py - scy[s], dz = pz - scz[s];
    const float d = dx * dx + dy * dy + dz * dz;
    if (d < d0)       { d2v = d1; i2 = i1; d1 = d0; i1 = i0; d0 = d; i0 = s; }
    else if (d < d1)  { d2v = d1; i2 = i1; d1 = d;  i1 = s; }
    else if (d < d2v) { d2v = d;  i2 = s; }
  }
  const float w0 = 1.f / fmaxf(d0, 1e-16f);
  const float w1 = 1.f / fmaxf(d1, 1e-16f);
  const float w2 = 1.f / fmaxf(d2v, 1e-16f);
  const float inv = 1.f / (w0 + w1 + w2);

  const float4* r0 = (const float4*)(x0 + (cb + i0) * 128);
  const float4* r1 = (const float4*)(x0 + (cb + i1) * 128);
  const float4* r2 = (const float4*)(x0 + (cb + i2) * 128);
  __builtin_prefetch(r0, 0, 1);
  __builtin_prefetch(r1, 0, 1);
  __builtin_prefetch(r2, 0, 1);
  _Float16* orow = up0h + gid * 128;
#pragma unroll 4
  for (int q = 0; q < 32; ++q) {
    const float4 a = r0[q], b = r1[q], c = r2[q];
    union { _Float16 h[4]; unsigned long long u; } pk;
    pk.h[0] = (_Float16)((w0 * a.x + w1 * b.x + w2 * c.x) * inv);
    pk.h[1] = (_Float16)((w0 * a.y + w1 * b.y + w2 * c.y) * inv);
    pk.h[2] = (_Float16)((w0 * a.z + w1 * b.z + w2 * c.z) * inv);
    pk.h[3] = (_Float16)((w0 * a.w + w1 * b.w + w2 * c.w) * inv);
    *(unsigned long long*)(orow + q * 4) = pk.u;
  }
}

// ---------------------------------------------------------------------------
// Kernel 5: fused head (128->64 relu, 64->64 relu, 64->10), WMMA.
// A-tile (64x128 f16, 16KB contiguous) staged into LDS by the TDM.
// ---------------------------------------------------------------------------
__global__ void head_kernel(const _Float16* __restrict__ up0h,
                            const _Float16* __restrict__ Wf,
                            const float* __restrict__ bf,
                            const _Float16* __restrict__ Wl0h,
                            const float* __restrict__ bl0,
                            const _Float16* __restrict__ Wl1h,
                            const float* __restrict__ bl1,
                            float* __restrict__ out) {
  __shared__ _Float16 sA[64 * 128];
  __shared__ _Float16 sH1[64 * 64];
  __shared__ _Float16 sH2[64 * 64];

  const int t       = threadIdx.x;
  const size_t row0 = (size_t)blockIdx.x * 64;

#if __has_builtin(__builtin_amdgcn_tensor_load_to_lds)
  if (t < 32) {  // one wave issues the tensor DMA (EXEC-independent, per-wave op)
    const unsigned long long ga = (unsigned long long)(size_t)(up0h + row0 * 128);
    const unsigned lds = (unsigned)(size_t)(void*)sA;
    u32x4 g0;
    g0[0] = 1u;                                         // count=1, user mode
    g0[1] = lds;                                        // D#.lds_addr
    g0[2] = (unsigned)(ga & 0xffffffffu);               // D#.global_addr[31:0]
    g0[3] = (unsigned)((ga >> 32) & 0x01ffffffu) | (2u << 30);  // addr[56:32] | type=2
    i32x8 g1;
    g1[0] = (1 << 16);          // workgroup_mask=0, data_size=1 (2 bytes)
    g1[1] = (int)(8192u << 16); // tensor_dim0 = 8192
    g1[2] = 0;
    g1[3] = (int)(8192u << 16); // tile_dim0 = 8192 (1-D tile, 16KB)
    g1[4] = 0;
    g1[5] = 8192;               // tensor_dim0_stride
    g1[6] = 0;
    g1[7] = 0;
    i32x4 gz = {0, 0, 0, 0};
#if defined(__clang_major__) && (__clang_major__ >= 23)
    i32x8 gz8 = {0, 0, 0, 0, 0, 0, 0, 0};
    __builtin_amdgcn_tensor_load_to_lds(g0, g1, gz, gz, gz8, 0);
#else
    __builtin_amdgcn_tensor_load_to_lds(g0, g1, gz, gz, 0);
#endif
    __builtin_amdgcn_s_wait_tensorcnt(0);
  }
#else
  for (int i = t; i < 64 * 128; i += 256) sA[i] = up0h[row0 * 128 + i];
#endif
  __syncthreads();

  const int wave = t >> 5, lane = t & 31;
  const int n = lane & 15, mb = (lane & 16) ? 8 : 0;

  for (int tt = wave; tt < 16; tt += 8) {
    const int mt = tt & 3, nt = tt >> 2;
    v8f acc = {};
#pragma unroll
    for (int kk = 0; kk < 4; ++kk) {
      v16h a = frag_a(sA, mt * 16, kk * 32, 128);
      v16h b = frag_bp(Wf, kk, nt, 4);
      acc = wmma_f16(a, b, acc);
    }
    const float bias = bf[nt * 16 + n];
#pragma unroll
    for (int i = 0; i < 8; ++i) {
      float v = acc[i] + bias;
      v = v > 0.f ? v : 0.f;
      sH1[(mt * 16 + mb + i) * 64 + nt * 16 + n] = (_Float16)v;
    }
  }
  __syncthreads();

  for (int tt = wave; tt < 16; tt += 8) {
    const int mt = tt & 3, nt = tt >> 2;
    v8f acc = {};
#pragma unroll
    for (int kk = 0; kk < 2; ++kk) {
      v16h a = frag_a(sH1, mt * 16, kk * 32, 64);
      v16h b = frag_bp(Wl0h, kk, nt, 4);
      acc = wmma_f16(a, b, acc);
    }
    const float bias = bl0[nt * 16 + n];
#pragma unroll
    for (int i = 0; i < 8; ++i) {
      float v = acc[i] + bias;
      v = v > 0.f ? v : 0.f;
      sH2[(mt * 16 + mb + i) * 64 + nt * 16 + n] = (_Float16)v;
    }
  }
  __syncthreads();

  if (wave < 4) {
    const int mt = wave;
    v8f acc = {};
#pragma unroll
    for (int kk = 0; kk < 2; ++kk) {
      v16h a = frag_a(sH2, mt * 16, kk * 32, 64);
      v16h b = frag_bp(Wl1h, kk, 0, 1);
      acc = wmma_f16(a, b, acc);
    }
    if (n < 10) {
      const float bias = bl1[n];
#pragma unroll
      for (int i = 0; i < 8; ++i)
        out[(row0 + mt * 16 + mb + i) * 10 + n] = acc[i] + bias;
    }
  }
}

// ---------------------------------------------------------------------------
// Workspace layout (bytes)
// ---------------------------------------------------------------------------
constexpr size_t OFF_CPOS = 0;
constexpr size_t OFF_NIDX = OFF_CPOS + (size_t)NB * NS0 * 3 * 4;
constexpr size_t OFF_ND2  = OFF_NIDX + (size_t)NB * NS0 * NKN * 4;
constexpr size_t OFF_X0   = OFF_ND2  + (size_t)NB * NS0 * NKN * 4;
constexpr size_t OFF_UP0H = OFF_X0   + (size_t)NB * NS0 * 128 * 4;
constexpr size_t OFF_WA   = OFF_UP0H + (size_t)NB * NP * 128 * 2;
constexpr size_t OFF_WB   = OFF_WA   + 1 * 4 * 512 * 2;
constexpr size_t OFF_WF   = OFF_WB   + 2 * 8 * 512 * 2;
constexpr size_t OFF_WL0  = OFF_WF   + 4 * 4 * 512 * 2;
constexpr size_t OFF_WL1  = OFF_WL0  + 2 * 4 * 512 * 2;

extern "C" void kernel_launch(void* const* d_in, const int* in_sizes, int n_in,
                              void* d_out, int out_size, void* d_ws, size_t ws_size,
                              hipStream_t stream) {
  (void)in_sizes; (void)n_in; (void)out_size; (void)ws_size;

  const float* pos    = (const float*)d_in[0];
  const float* W_sa0a = (const float*)d_in[2];
  const float* b_sa0a = (const float*)d_in[3];
  const float* W_sa0b = (const float*)d_in[4];
  const float* b_sa0b = (const float*)d_in[5];
  const float* W_fp0  = (const float*)d_in[12];
  const float* b_fp0  = (const float*)d_in[13];
  const float* W_l0   = (const float*)d_in[14];
  const float* b_l0   = (const float*)d_in[15];
  const float* W_l1   = (const float*)d_in[16];
  const float* b_l1   = (const float*)d_in[17];

  char* ws        = (char*)d_ws;
  float*     cpos = (float*)(ws + OFF_CPOS);
  int*       nidx = (int*)  (ws + OFF_NIDX);
  float*     nd2  = (float*)(ws + OFF_ND2);
  float*     x0   = (float*)(ws + OFF_X0);
  _Float16*  up0h = (_Float16*)(ws + OFF_UP0H);
  _Float16*  Wa   = (_Float16*)(ws + OFF_WA);
  _Float16*  Wb   = (_Float16*)(ws + OFF_WB);
  _Float16*  Wf   = (_Float16*)(ws + OFF_WF);
  _Float16*  Wl0h = (_Float16*)(ws + OFF_WL0);
  _Float16*  Wl1h = (_Float16*)(ws + OFF_WL1);
  float*     out  = (float*)d_out;

  convert_weights_kernel<<<1, 256, 0, stream>>>(W_sa0a, W_sa0b, W_fp0, W_l0, W_l1,
                                                Wa, Wb, Wf, Wl0h, Wl1h);
  fps_kernel<<<NB, 1024, 0, stream>>>(pos, cpos);
  ballquery_kernel<<<NB * NS0, 256, 0, stream>>>(pos, cpos, nidx, nd2);
  sa0_mlp_kernel<<<NB * NS0, 256, 0, stream>>>(pos, cpos, nidx, nd2,
                                               Wa, Wb, b_sa0a, b_sa0b, x0);
  knn_interp_kernel<<<(NB * NP) / 256, 256, 0, stream>>>(pos, cpos, x0, up0h);
  head_kernel<<<(NB * NP) / 64, 256, 0, stream>>>(up0h, Wf, b_fp0, Wl0h, b_l0,
                                                  Wl1h, b_l1, out);
}
